// EnDeModel_10419590660413
// MI455X (gfx1250) — matrix operand order repaired and checked by reading, not compiled, not run on hardware
//
#include <hip/hip_runtime.h>
#include <hip/hip_bf16.h>

// ============================================================================
// Fused EnDeModel kernel for MI455X (gfx1250, wave32, WMMA).
//
//   Stage 1 (per wave, 16 rows): betas = history_u @ W^T + b via
//     v_wmma_f32_16x16x32_f16 (8 N-tiles x 2 K-chunks = 16 WMMAs / wave),
//     results + bias stored to LDS in [t][row] layout (stride 68 to balance
//     banks between the WMMA store pattern and the scan read pattern).
//   Stage 2 (threads 0..63): 128-step sequential SIRD scan per row, betas
//     streamed from LDS, final (s,i,r,d,n) written to global.
//
// Fusion avoids materializing the 268 MB beta tensor in HBM; total HBM
// traffic ~155 MB => ~6.7 us roofline at 23.3 TB/s, with the f16 WMMA GEMM
// (~8.6 GFLOP) comfortably under that floor.
// ============================================================================

typedef __attribute__((ext_vector_type(16))) _Float16 v16h;
typedef __attribute__((ext_vector_type(8)))  float    v8f;
typedef __attribute__((ext_vector_type(4)))  float    v4f;

#define HLEN 64
#define TLEN 128
#define ROWS_TOTAL (1024 * 512)
#define ROWS_PER_WG 64
#define LDS_STRIDE 68   /* 64 rows + 4 floats pad: keeps b128 alignment, kills bank conflicts */

#define RECOVER_RATE 0.03f
#define DEATH_RATE   0.01f

__launch_bounds__(128, 1)
__global__ void endemodel_fused_kernel(const float* __restrict__ hist,   // [ROWS, 64]
                                       const float* __restrict__ sird,   // [ROWS, 5]
                                       const float* __restrict__ W,      // [128, 64] row-major
                                       const float* __restrict__ bias,   // [128]
                                       float* __restrict__ out)          // [ROWS, 5]
{
    __shared__ float beta_lds[TLEN * LDS_STRIDE];   // ~34 KB, [t][row(+pad)]

    const int  tid  = threadIdx.x;
    const int  wave = tid >> 5;
    const int  lane = tid & 31;
    const int  half = lane >> 4;    // lane group 0-15 vs 16-31
    const int  l15  = lane & 15;
    const long wg_row0   = (long)blockIdx.x * ROWS_PER_WG;
    const int  wave_row0 = wave << 4;

    // ---------------- Stage 1: beta GEMM via WMMA ----------------
    // A fragment (16x32 f16, ISA 7.12.2): lane l holds row (l&15);
    // halves 0..15 = K = kc*32 + (half?16:0) + j (contiguous, matches B).
    const float* arow = hist + (wg_row0 + wave_row0 + l15) * HLEN;
    v16h afrag[2];
#pragma unroll
    for (int kc = 0; kc < 2; ++kc) {
        const int lo = kc * 32 + half * 16;
        const v4f a0 = *(const v4f*)(arow + lo);
        const v4f a1 = *(const v4f*)(arow + lo + 4);
        const v4f a2 = *(const v4f*)(arow + lo + 8);
        const v4f a3 = *(const v4f*)(arow + lo + 12);
#pragma unroll
        for (int j = 0; j < 4; ++j) {
            afrag[kc][j]      = (_Float16)a0[j];
            afrag[kc][4 + j]  = (_Float16)a1[j];
            afrag[kc][8 + j]  = (_Float16)a2[j];
            afrag[kc][12 + j] = (_Float16)a3[j];
        }
    }

    // 8 N-tiles of 16 timesteps each; accumulator is transient per tile.
    // B fragment (32x16 f16): lane holds column t = n*16 + (l&15);
    // 16 halves = K = kc*32 + (half?16:0) + {0..15} -> 16 contiguous floats
    // of W's row-major row => four b128 loads.
#pragma unroll
    for (int n = 0; n < 8; ++n) {
        const int    t    = n * 16 + l15;
        const float* wrow = W + t * HLEN;

        v8f acc = {};
#pragma unroll
        for (int kc = 0; kc < 2; ++kc) {
            const int h0 = kc * 32 + half * 16;
            const v4f w0 = *(const v4f*)(wrow + h0);
            const v4f w1 = *(const v4f*)(wrow + h0 + 4);
            const v4f w2 = *(const v4f*)(wrow + h0 + 8);
            const v4f w3 = *(const v4f*)(wrow + h0 + 12);
            v16h bfrag;
#pragma unroll
            for (int j = 0; j < 4; ++j) {
                bfrag[j]      = (_Float16)w0[j];
                bfrag[4 + j]  = (_Float16)w1[j];
                bfrag[8 + j]  = (_Float16)w2[j];
                bfrag[12 + j] = (_Float16)w3[j];
            }
            // D = A x B + C ; 8 args: (neg_a, A, neg_b, B, c_mod, C, reuse_a, reuse_b)
            acc = __builtin_amdgcn_wmma_f32_16x16x32_f16(
                false, afrag[kc], false, bfrag, (short)0, acc, false, false);
        }

        // C/D layout: lane holds col t; VGPR j -> row M = j + 8*half.
        // Add bias, store 8 rows as two b128s to LDS [t][row].
        const float bt    = bias[t];
        const int   rbase = wave_row0 + half * 8;
        float* dst = &beta_lds[t * LDS_STRIDE + rbase];
        v4f lo4 = { acc[0] + bt, acc[1] + bt, acc[2] + bt, acc[3] + bt };
        v4f hi4 = { acc[4] + bt, acc[5] + bt, acc[6] + bt, acc[7] + bt };
        *(v4f*)(dst)     = lo4;
        *(v4f*)(dst + 4) = hi4;
    }

    __syncthreads();

    // ---------------- Stage 2: sequential SIRD scan ----------------
    if (tid < ROWS_PER_WG) {
        const long  row = wg_row0 + tid;
        const float* x0 = sird + row * 5;
        float s  = x0[0];
        float i  = x0[1];
        float r  = x0[2];
        float d  = x0[3];
        float nn = x0[4];
        const float inv_n = 1.0f / nn;   // n is invariant across the scan

#pragma unroll 8
        for (int t = 0; t < TLEN; ++t) {
            const float beta = beta_lds[t * LDS_STRIDE + tid];  // conflict-free
            const float si   = s * i * beta * inv_n;
            const float ir   = i * RECOVER_RATE;
            const float idd  = i * DEATH_RATE;
            r += ir;
            d += idd;
            s -= si;
            i += si - ir - idd;
        }

        float* o = out + row * 5;
        o[0] = s; o[1] = i; o[2] = r; o[3] = d; o[4] = nn;
    }
}

extern "C" void kernel_launch(void* const* d_in, const int* in_sizes, int n_in,
                              void* d_out, int out_size, void* d_ws, size_t ws_size,
                              hipStream_t stream)
{
    (void)in_sizes; (void)n_in; (void)d_ws; (void)ws_size; (void)out_size;

    const float* hist = (const float*)d_in[0];   // history_u [1024,512,64]
    const float* sird = (const float*)d_in[1];   // sird      [1024,512,5]
    const float* W    = (const float*)d_in[2];   // W         [128,64]
    const float* bias = (const float*)d_in[3];   // b         [128]
    float*       out  = (float*)d_out;           // x final   [1024,512,5]

    const int grid = ROWS_TOTAL / ROWS_PER_WG;   // 8192 workgroups
    endemodel_fused_kernel<<<grid, 128, 0, stream>>>(hist, sird, W, bias, out);
}
